// SigLipLoss_7894149890372
// MI455X (gfx1250) — compile-verified, matrix operand order checked by reading, not run on hardware
//
#include <hip/hip_runtime.h>

typedef _Float16 v16h __attribute__((ext_vector_type(16)));
typedef _Float16 v8h  __attribute__((ext_vector_type(8)));
typedef float    v8f  __attribute__((ext_vector_type(8)));
typedef float    v4f  __attribute__((ext_vector_type(4)));

#define TILE_M 256            // block tile M
#define TILE_N 256            // block tile N
#define TILE_K 32             // K slice per stage
#define DIM    512
#define NTHREADS 512          // 16 waves (wave32), wave grid 4(M) x 4(N)

// ---------------------------------------------------------------------------
// Shared device helpers
// ---------------------------------------------------------------------------
__device__ __forceinline__ v16h load_frag_lds(const _Float16* base, int row0,
                                              int r, int h)
{
    // 16-bit operand fragment per ISA layout:
    // lane<16 (h=0): K {0..7} and {16..23}; lane>=16 (h=1): K {8..15},{24..31}.
    const _Float16* p = base + (size_t)(row0 + r) * TILE_K + h * 8;
    v8h lo = *(const v8h*)(p);
    v8h hi = *(const v8h*)(p + 16);
    v16h out;
#pragma unroll
    for (int e = 0; e < 8; ++e) { out[e] = lo[e]; out[8 + e] = hi[e]; }
    return out;
}

__device__ __forceinline__ float logsig_neg(float x)
{
    // -log_sigmoid(x), numerically stable
    return -(fminf(x, 0.0f) - __logf(1.0f + __expf(-fabsf(x))));
}

// ---------------------------------------------------------------------------
// fp32 -> f16 pre-conversion (one pass, HBM-bandwidth bound, ~4us)
// ---------------------------------------------------------------------------
__launch_bounds__(256)
__global__ void cvt_f32_to_f16_kernel(const float* __restrict__ src,
                                      _Float16* __restrict__ dst, int n8)
{
    int i = blockIdx.x * blockDim.x + threadIdx.x;
    if (i >= n8) return;
    v4f a = *(const v4f*)(src + (size_t)i * 8);
    v4f b = *(const v4f*)(src + (size_t)i * 8 + 4);
    v8h hv;
#pragma unroll
    for (int e = 0; e < 4; ++e) { hv[e] = (_Float16)a[e]; hv[4 + e] = (_Float16)b[e]; }
    *(v8h*)(dst + (size_t)i * 8) = hv;
}

// ---------------------------------------------------------------------------
// Fast path: f16 inputs, async Global->LDS DMA (ASYNCcnt), triple-buffered
// ---------------------------------------------------------------------------
__launch_bounds__(NTHREADS)
__global__ void siglip_tile_async_kernel(const _Float16* __restrict__ imgh,
                                         const _Float16* __restrict__ txth,
                                         const float* __restrict__ scale_p,
                                         const float* __restrict__ bias_p,
                                         float* __restrict__ partial)
{
    __shared__ __align__(16) _Float16 Ash[3][TILE_M * TILE_K]; // 3 x 16 KB
    __shared__ __align__(16) _Float16 Bsh[3][TILE_N * TILE_K]; // 3 x 16 KB
    __shared__ float red[NTHREADS];

    const int tid  = threadIdx.x;
    const int lane = tid & 31;
    const int wave = tid >> 5;   // 0..15
    const int wm   = wave >> 2;  // 0..3 -> M offset wm*64
    const int wn   = wave & 3;   // 0..3 -> N offset wn*64
    const int r    = lane & 15;
    const int h    = lane >> 4;
    const int m0   = blockIdx.x * TILE_M;
    const int n0   = blockIdx.y * TILE_N;

    const float scale = *scale_p;
    const float bias  = *bias_p;

    v8f acc[4][4] = {};

    // Issue 4 async 16B Global->LDS copies per thread for one K-slice.
    // Flat 16B chunks: 0..1023 -> A tile (256 rows x 4 chunks of 8 halves),
    //                  1024..2047 -> B tile.
    auto issue_stage = [&](int k0, int b) {
#pragma unroll
        for (int c = 0; c < 4; ++c) {
            int flat = tid + c * NTHREADS;
            bool isB = flat >= 1024;
            int  f   = isB ? (flat - 1024) : flat;
            int  row = f >> 2;
            int  col = (f & 3) * 8;
            const _Float16* g = (isB ? (txth + (size_t)(n0 + row) * DIM)
                                     : (imgh + (size_t)(m0 + row) * DIM)) + k0 + col;
            // generic LDS address: low 32 bits == byte offset in LDS
            unsigned lds = (unsigned)(size_t)((isB ? &Bsh[b][0] : &Ash[b][0]) + f * 8);
            asm volatile("global_load_async_to_lds_b128 %0, %1, off"
                         :: "v"(lds), "v"((unsigned long long)(size_t)g)
                         : "memory");
        }
    };

    const int nkt = DIM / TILE_K; // 16

    issue_stage(0, 0);

    for (int kt = 0; kt < nkt; ++kt) {
        const int b = kt % 3;
        if (kt + 1 < nkt) {
            issue_stage((kt + 1) * TILE_K, (kt + 1) % 3); // overlap with WMMAs
            asm volatile("s_wait_asynccnt 0x4" ::: "memory"); // stage kt landed
        } else {
            asm volatile("s_wait_asynccnt 0x0" ::: "memory");
        }
        __syncthreads();

        v16h bf[4];
#pragma unroll
        for (int ni = 0; ni < 4; ++ni)
            bf[ni] = load_frag_lds(&Bsh[b][0], wn * 64 + ni * 16, r, h);

#pragma unroll
        for (int mi = 0; mi < 4; ++mi) {
            v16h af = load_frag_lds(&Ash[b][0], wm * 64 + mi * 16, r, h);
#pragma unroll
            for (int ni = 0; ni < 4; ++ni)
                acc[mi][ni] = __builtin_amdgcn_wmma_f32_16x16x32_f16(
                    false, af, false, bf[ni], (short)0, acc[mi][ni],
                    false, false);
        }
    }

    // Fused SigLIP epilogue
    float lsum = 0.0f;
#pragma unroll
    for (int mi = 0; mi < 4; ++mi) {
#pragma unroll
        for (int ni = 0; ni < 4; ++ni) {
            const int gn = n0 + wn * 64 + ni * 16 + r;
#pragma unroll
            for (int e = 0; e < 8; ++e) {
                const int gm = m0 + wm * 64 + mi * 16 + h * 8 + e;
                float x = scale * acc[mi][ni][e] + bias;
                if (gm != gn) x = -x;
                lsum += logsig_neg(x);
            }
        }
    }

    red[tid] = lsum;
    __syncthreads();
#pragma unroll
    for (int off = NTHREADS / 2; off > 0; off >>= 1) {
        if (tid < off) red[tid] += red[tid + off];
        __syncthreads();
    }
    if (tid == 0) partial[blockIdx.x * gridDim.y + blockIdx.y] = red[0];
}

// ---------------------------------------------------------------------------
// Fallback path (ws too small): fp32 inputs, on-the-fly convert (round-2 kernel)
// ---------------------------------------------------------------------------
__launch_bounds__(NTHREADS)
__global__ void siglip_tile_kernel(const float* __restrict__ img,
                                   const float* __restrict__ txt,
                                   const float* __restrict__ scale_p,
                                   const float* __restrict__ bias_p,
                                   float* __restrict__ partial)
{
    __shared__ __align__(16) _Float16 Ash[2][TILE_M * TILE_K];
    __shared__ __align__(16) _Float16 Bsh[2][TILE_N * TILE_K];
    __shared__ float red[NTHREADS];

    const int tid  = threadIdx.x;
    const int lane = tid & 31;
    const int wave = tid >> 5;
    const int wm   = wave >> 2;
    const int wn   = wave & 3;
    const int r    = lane & 15;
    const int h    = lane >> 4;
    const int m0   = blockIdx.x * TILE_M;
    const int n0   = blockIdx.y * TILE_N;

    const float scale = *scale_p;
    const float bias  = *bias_p;

    v8f acc[4][4] = {};
    v4f st[4][2];

    auto load_stage = [&](int k0) {
#pragma unroll
        for (int c = 0; c < 4; ++c) {
            int flat8 = tid + c * NTHREADS;
            bool isB  = flat8 >= 1024;
            int  f    = isB ? (flat8 - 1024) : flat8;
            int  row  = f >> 2;
            int  col  = (f & 3) * 8;
            const float* src = (isB ? (txt + (size_t)(n0 + row) * DIM)
                                    : (img + (size_t)(m0 + row) * DIM)) + k0 + col;
            st[c][0] = *(const v4f*)(src);
            st[c][1] = *(const v4f*)(src + 4);
        }
    };

    auto store_stage = [&](int s) {
#pragma unroll
        for (int c = 0; c < 4; ++c) {
            int flat8 = tid + c * NTHREADS;
            bool isB  = flat8 >= 1024;
            int  f    = isB ? (flat8 - 1024) : flat8;
            _Float16* dst = (isB ? &Bsh[s][0] : &Ash[s][0]) + f * 8;
            v8h hv;
#pragma unroll
            for (int e = 0; e < 4; ++e) hv[e]     = (_Float16)st[c][0][e];
#pragma unroll
            for (int e = 0; e < 4; ++e) hv[4 + e] = (_Float16)st[c][1][e];
            *(v8h*)dst = hv;
        }
    };

    const int nkt = DIM / TILE_K;

    load_stage(0);
    store_stage(0);
    __syncthreads();

    for (int kt = 0; kt < nkt; ++kt) {
        const int s = kt & 1;
        if (kt + 1 < nkt) load_stage((kt + 1) * TILE_K);

        v16h bf[4];
#pragma unroll
        for (int ni = 0; ni < 4; ++ni)
            bf[ni] = load_frag_lds(&Bsh[s][0], wn * 64 + ni * 16, r, h);

#pragma unroll
        for (int mi = 0; mi < 4; ++mi) {
            v16h af = load_frag_lds(&Ash[s][0], wm * 64 + mi * 16, r, h);
#pragma unroll
            for (int ni = 0; ni < 4; ++ni)
                acc[mi][ni] = __builtin_amdgcn_wmma_f32_16x16x32_f16(
                    false, af, false, bf[ni], (short)0, acc[mi][ni],
                    false, false);
        }

        if (kt + 1 < nkt) {
            store_stage(s ^ 1);
            __syncthreads();
        }
    }

    float lsum = 0.0f;
#pragma unroll
    for (int mi = 0; mi < 4; ++mi) {
#pragma unroll
        for (int ni = 0; ni < 4; ++ni) {
            const int gn = n0 + wn * 64 + ni * 16 + r;
#pragma unroll
            for (int e = 0; e < 8; ++e) {
                const int gm = m0 + wm * 64 + mi * 16 + h * 8 + e;
                float x = scale * acc[mi][ni][e] + bias;
                if (gm != gn) x = -x;
                lsum += logsig_neg(x);
            }
        }
    }

    red[tid] = lsum;
    __syncthreads();
#pragma unroll
    for (int off = NTHREADS / 2; off > 0; off >>= 1) {
        if (tid < off) red[tid] += red[tid + off];
        __syncthreads();
    }
    if (tid == 0) partial[blockIdx.x * gridDim.y + blockIdx.y] = red[0];
}

// ---------------------------------------------------------------------------
// Final deterministic reduction
// ---------------------------------------------------------------------------
__launch_bounds__(256)
__global__ void siglip_reduce_kernel(const float* __restrict__ partial, int n,
                                     float* __restrict__ out, float invN)
{
    __shared__ float red[256];
    float s = 0.0f;
    for (int i = threadIdx.x; i < n; i += 256) s += partial[i]; // fixed order
    red[threadIdx.x] = s;
    __syncthreads();
#pragma unroll
    for (int off = 128; off > 0; off >>= 1) {
        if (threadIdx.x < off) red[threadIdx.x] += red[threadIdx.x + off];
        __syncthreads();
    }
    if (threadIdx.x == 0) out[0] = red[0] * invN;
}

extern "C" void kernel_launch(void* const* d_in, const int* in_sizes, int n_in,
                              void* d_out, int out_size, void* d_ws, size_t ws_size,
                              hipStream_t stream)
{
    const float* img     = (const float*)d_in[0];
    const float* txt     = (const float*)d_in[1];
    const float* scale_p = (const float*)d_in[2];
    const float* bias_p  = (const float*)d_in[3];
    float*       out     = (float*)d_out;

    const int N = in_sizes[0] / DIM; // 16384
    dim3 grid(N / TILE_M, N / TILE_N);
    const int nPartial = (N / TILE_M) * (N / TILE_N);

    const size_t halfBytes = (size_t)N * DIM * sizeof(_Float16); // 16 MB each
    const size_t needFast  = 2 * halfBytes + (size_t)nPartial * sizeof(float);

    if (ws_size >= needFast) {
        _Float16* imgh    = (_Float16*)d_ws;
        _Float16* txth    = (_Float16*)((char*)d_ws + halfBytes);
        float*    partial = (float*)((char*)d_ws + 2 * halfBytes);

        const int n8 = N * DIM / 8;
        cvt_f32_to_f16_kernel<<<(n8 + 255) / 256, 256, 0, stream>>>(img, imgh, n8);
        cvt_f32_to_f16_kernel<<<(n8 + 255) / 256, 256, 0, stream>>>(txt, txth, n8);

        siglip_tile_async_kernel<<<grid, NTHREADS, 0, stream>>>(
            imgh, txth, scale_p, bias_p, partial);
        siglip_reduce_kernel<<<1, 256, 0, stream>>>(partial, nPartial, out,
                                                    1.0f / (float)N);
    } else {
        float* partial = (float*)d_ws; // 16 KB
        siglip_tile_kernel<<<grid, NTHREADS, 0, stream>>>(
            img, txt, scale_p, bias_p, partial);
        siglip_reduce_kernel<<<1, 256, 0, stream>>>(partial, nPartial, out,
                                                    1.0f / (float)N);
    }
}
